// PhaseTrackerStatic_16286515986739
// MI455X (gfx1250) — matrix operand order repaired and checked by reading, not compiled, not run on hardware
//
#include <hip/hip_runtime.h>
#include <hip/hip_bf16.h>
#include <math.h>

// ---------------- problem constants (from reference) ----------------
#define NROWS 8192   // N detections_t
#define MROWS 8192   // M detections_t1
#define DDIM  4
#define HDIM  64
#define NOSC  28
#define KPAD  64     // 28 cos + 28 sin + 8 zero pad
#define TWO_PI_F   6.283185307179586f
#define INV_2PI_F  0.15915494309189535f
#define DT_F       0.01f
#define THRESH     0.3f

typedef _Float16 half8  __attribute__((ext_vector_type(8)));
typedef _Float16 half16 __attribute__((ext_vector_type(16)));
typedef float    float8 __attribute__((ext_vector_type(8)));

// order-preserving float->uint key (monotonic for all finite floats)
__device__ __forceinline__ unsigned int ord_encode(float v) {
    unsigned int u = __float_as_uint(v);
    return (u & 0x80000000u) ? ~u : (u | 0x80000000u);
}
__device__ __forceinline__ float ord_decode(unsigned int ordv) {
    unsigned int u = (ordv & 0x80000000u) ? (ordv ^ 0x80000000u) : ~ordv;
    return __uint_as_float(u);
}
// u64 shuffle-xor via two 32-bit shuffles (portable across HIP versions)
__device__ __forceinline__ unsigned long long shfl_xor_u64(unsigned long long k, int mask) {
    int lo = (int)(unsigned int)(k & 0xFFFFFFFFull);
    int hi = (int)(unsigned int)(k >> 32);
    lo = __shfl_xor(lo, mask, 32);
    hi = __shfl_xor(hi, mask, 32);
    return ((unsigned long long)(unsigned int)hi << 32) | (unsigned int)lo;
}

// =====================================================================
// Kernel 1: phase MLP encode + (for set t) 5-step phase advance,
// emit f16 [cos | sin | 0pad] rows (K=64) for the WMMA GEMM.
// Also initializes the per-row argmax key array.
// =====================================================================
__global__ __launch_bounds__(256) void pts_encode_kernel(
    const float* __restrict__ det_t, const float* __restrict__ det_t1,
    const float* __restrict__ W1, const float* __restrict__ b1,
    const float* __restrict__ W2, const float* __restrict__ b2,
    const float* __restrict__ freqs,
    _Float16* __restrict__ Amat, _Float16* __restrict__ Bmat,
    unsigned long long* __restrict__ rowbest)
{
    int row = blockIdx.x * blockDim.x + threadIdx.x;
    if (row >= NROWS + MROWS) return;
    bool isT = row < NROWS;
    if (isT) rowbest[row] = 0ull;   // minimal key; real values always larger
    const float* x = isT ? (det_t + (size_t)row * DDIM)
                         : (det_t1 + (size_t)(row - NROWS) * DDIM);
    float x0 = x[0], x1 = x[1], x2 = x[2], x3 = x[3];

    float acc[NOSC];
#pragma unroll
    for (int o = 0; o < NOSC; ++o) acc[o] = b2[o];

    for (int h = 0; h < HDIM; ++h) {
        float hv = x0 * W1[0 * HDIM + h] + x1 * W1[1 * HDIM + h]
                 + x2 * W1[2 * HDIM + h] + x3 * W1[3 * HDIM + h] + b1[h];
        hv = fmaxf(hv, 0.0f);
        const float* w2r = W2 + (size_t)h * NOSC;
#pragma unroll
        for (int o = 0; o < NOSC; ++o) acc[o] += hv * w2r[o];
    }

    _Float16* dst = isT ? (Amat + (size_t)row * KPAD)
                        : (Bmat + (size_t)(row - NROWS) * KPAD);
#pragma unroll
    for (int o = 0; o < NOSC; ++o) {
        float ph = acc[o];
        ph = ph - floorf(ph * INV_2PI_F) * TWO_PI_F;   // raw % 2pi
        if (isT) {
            float step = TWO_PI_F * freqs[o] * DT_F;
#pragma unroll
            for (int s = 0; s < 5; ++s) {
                ph += step;
                ph = ph - floorf(ph * INV_2PI_F) * TWO_PI_F;
            }
        }
        dst[o]        = (_Float16)cosf(ph);
        dst[NOSC + o] = (_Float16)sinf(ph);
    }
#pragma unroll
    for (int k = 2 * NOSC; k < KPAD; ++k) dst[k] = (_Float16)0.0f;
}

// =====================================================================
// Kernel 2: sim = scale * A @ B^T via v_wmma_f32_16x16x32_f16,
// with FUSED per-row max/argmax (avoids re-reading 256MB of sim).
// Block = 256 threads (8 waves), 128x128 output tile.
// Wave (wr in 0..3, wc in 0..1) computes 32 rows x 64 cols = 2x4 WMMA tiles.
// K = 64 -> two k-steps of 32. NT stores (256MB > 192MB L2).
// =====================================================================
__global__ __launch_bounds__(256) void pts_simgemm_kernel(
    const _Float16* __restrict__ A, const _Float16* __restrict__ B,
    float* __restrict__ C, unsigned long long* __restrict__ rowbest)
{
    const int lane  = threadIdx.x & 31;
    const int wave  = threadIdx.x >> 5;
    const int wr    = wave >> 1;
    const int wc    = wave & 1;
    const int mBase = blockIdx.y * 128 + wr * 32;
    const int nBase = blockIdx.x * 128 + wc * 64;
    const int l15   = lane & 15;
    const int khalf = lane >> 4;

    float8 acc[2][4];
#pragma unroll
    for (int i = 0; i < 2; ++i)
#pragma unroll
        for (int j = 0; j < 4; ++j)
#pragma unroll
            for (int r = 0; r < 8; ++r) acc[i][j][r] = 0.0f;

#pragma unroll
    for (int k0 = 0; k0 < KPAD; k0 += 32) {
        // A fragments: 16-bit A 16x32 layout:
        //   lane l (m = l&15, khalf = l>>4): elems 0..7  = K k0+khalf*8+0..7
        //                                    elems 8..15 = K k0+16+khalf*8+0..7
        half16 afrag[2];
#pragma unroll
        for (int t = 0; t < 2; ++t) {
            const _Float16* p = A + (size_t)(mBase + t * 16 + l15) * KPAD
                                  + k0 + khalf * 8;
            union { half16 v; half8 h[2]; } u;
            u.h[0] = *(const half8*)(p);
            u.h[1] = *(const half8*)(p + 16);
            afrag[t] = u.v;
        }
        // B fragments: lane l (n = l&15, khalf = l>>4): elems 0..7 = K k0+khalf*8+0..7, elems 8..15 = K k0+16+khalf*8+0..7
        half16 bfrag[4];
#pragma unroll
        for (int t = 0; t < 4; ++t) {
            const _Float16* p = B + (size_t)(nBase + t * 16 + l15) * KPAD
                                  + k0 + khalf * 8;
            union { half16 v; half8 h[2]; } u;
            u.h[0] = *(const half8*)(p);
            u.h[1] = *(const half8*)(p + 16);
            bfrag[t] = u.v;
        }
#pragma unroll
        for (int i = 0; i < 2; ++i)
#pragma unroll
            for (int j = 0; j < 4; ++j)
                acc[i][j] = __builtin_amdgcn_wmma_f32_16x16x32_f16(
                    false, afrag[i], false, bfrag[j],
                    (short)0, acc[i][j], false, false);
    }

    // constant normalization: na = nb = sqrt(28)+eps for every row
    const float na = sqrtf((float)NOSC) + 1e-6f;
    const float scale = 1.0f / (na * na);

    // C layout (f32 16x16): lanes 0-15: N=lane, M=r; lanes 16-31: N=lane-16, M=8+r
    // Epilogue: NT-store scaled values AND reduce per-row (max, first argmax).
#pragma unroll
    for (int i = 0; i < 2; ++i) {
#pragma unroll
        for (int r = 0; r < 8; ++r) {
            const int rowIdx = mBase + i * 16 + khalf * 8 + r;
            float* crow = C + (size_t)rowIdx * MROWS;
            float best = -INFINITY;
            int   bcol = 0;
#pragma unroll
            for (int j = 0; j < 4; ++j) {
                const int col = nBase + j * 16 + l15;
                float v = acc[i][j][r] * scale;
                __builtin_nontemporal_store(v, crow + col);
                if (v > best) { best = v; bcol = col; }   // strict > keeps first col
            }
            // key: larger value wins; on ties larger (M-1-col) => smaller col wins
            unsigned long long key =
                ((unsigned long long)ord_encode(best) << 32) |
                (unsigned long long)(unsigned int)(MROWS - 1 - bcol);
#pragma unroll
            for (int m = 1; m < 16; m <<= 1) {           // reduce across l15 group
                unsigned long long o = shfl_xor_u64(key, m);
                if (o > key) key = o;
            }
            if (l15 == 0)
                atomicMax(&rowbest[rowIdx], key);
        }
    }
}

// =====================================================================
// Kernel 3: decode packed (value, argmax) keys
// =====================================================================
__global__ __launch_bounds__(256) void pts_decode_kernel(
    const unsigned long long* __restrict__ rowbest,
    float* __restrict__ maxs, int* __restrict__ idxs)
{
    const int i = blockIdx.x * 256 + threadIdx.x;
    if (i >= NROWS) return;
    unsigned long long k = rowbest[i];
    maxs[i] = ord_decode((unsigned int)(k >> 32));
    idxs[i] = MROWS - 1 - (int)(unsigned int)(k & 0xFFFFFFFFull);
}

// =====================================================================
// Kernel 4: stable descending rank (reproduces argsort(-maxs) order),
// plus winner-array init for the claim pass.
// =====================================================================
__global__ __launch_bounds__(256) void pts_rank_kernel(
    const float* __restrict__ maxs, int* __restrict__ rank,
    int* __restrict__ winner)
{
    const int i = blockIdx.x * 256 + threadIdx.x;
    if (i >= NROWS) return;
    const float si = maxs[i];
    int rk = 0;
    for (int j = 0; j < NROWS; ++j) {
        float sj = maxs[j];
        rk += (sj > si) || (sj == si && j < i);
    }
    rank[i] = rk;
    winner[i] = 0x7FFFFFFF;
}

// =====================================================================
// Kernel 5: claim — smallest rank wins each target column.
// (Greedy loop is equivalent: each row targets one column; "used" is only
//  ever set by a winner, so winner of col j = min-rank candidate with
//  sim > threshold.)
// =====================================================================
__global__ __launch_bounds__(256) void pts_claim_kernel(
    const float* __restrict__ maxs, const int* __restrict__ idxs,
    const int* __restrict__ rank, int* __restrict__ winner)
{
    const int i = blockIdx.x * 256 + threadIdx.x;
    if (i >= NROWS) return;
    if (maxs[i] > THRESH) atomicMin(&winner[idxs[i]], rank[i]);
}

// =====================================================================
// Kernel 6: emit matches (as float, per harness output convention)
// =====================================================================
__global__ __launch_bounds__(256) void pts_match_kernel(
    const float* __restrict__ maxs, const int* __restrict__ idxs,
    const int* __restrict__ rank, const int* __restrict__ winner,
    float* __restrict__ out)
{
    const int i = blockIdx.x * 256 + threadIdx.x;
    if (i >= NROWS) return;
    int m = -1;
    if (maxs[i] > THRESH && winner[idxs[i]] == rank[i]) m = idxs[i];
    out[i] = (float)m;
}

// =====================================================================
extern "C" void kernel_launch(void* const* d_in, const int* in_sizes, int n_in,
                              void* d_out, int out_size, void* d_ws, size_t ws_size,
                              hipStream_t stream)
{
    const float* det_t  = (const float*)d_in[0];
    const float* det_t1 = (const float*)d_in[1];
    const float* W1     = (const float*)d_in[2];
    const float* b1     = (const float*)d_in[3];
    const float* W2     = (const float*)d_in[4];
    const float* b2     = (const float*)d_in[5];
    const float* freqs  = (const float*)d_in[6];

    float* out     = (float*)d_out;            // [0..N)       matches (float)
    float* sim     = out + NROWS;              // [N..N+N*M)   similarity matrix

    char* ws = (char*)d_ws;
    _Float16* Amat = (_Float16*)ws;                                   // 1 MB
    _Float16* Bmat = (_Float16*)(ws + (1u << 20));                    // 1 MB
    float* maxs    = (float*)(ws + (2u << 20));                       // 32 KB
    int*   idxs    = (int*)  (ws + (2u << 20) + 32768);               // 32 KB
    int*   rank    = (int*)  (ws + (2u << 20) + 65536);               // 32 KB
    int*   winner  = (int*)  (ws + (2u << 20) + 98304);               // 32 KB
    unsigned long long* rowbest =
        (unsigned long long*)(ws + (2u << 20) + 131072);              // 64 KB

    // 1) encode both detection sets -> f16 [cos|sin|pad] rows; init rowbest
    pts_encode_kernel<<<dim3((NROWS + MROWS + 255) / 256), dim3(256), 0, stream>>>(
        det_t, det_t1, W1, b1, W2, b2, freqs, Amat, Bmat, rowbest);

    // 2) WMMA GEMM with fused row max/argmax: sim = scale * A @ B^T
    pts_simgemm_kernel<<<dim3(MROWS / 128, NROWS / 128), dim3(256), 0, stream>>>(
        Amat, Bmat, sim, rowbest);

    // 3) decode (max, argmax) keys
    pts_decode_kernel<<<dim3(NROWS / 256), dim3(256), 0, stream>>>(rowbest, maxs, idxs);

    // 4) ranks + winner init
    pts_rank_kernel<<<dim3(NROWS / 256), dim3(256), 0, stream>>>(maxs, rank, winner);

    // 5) claim columns
    pts_claim_kernel<<<dim3(NROWS / 256), dim3(256), 0, stream>>>(maxs, idxs, rank, winner);

    // 6) emit matches
    pts_match_kernel<<<dim3(NROWS / 256), dim3(256), 0, stream>>>(maxs, idxs, rank, winner, out);
}
